// SemanticVQVAE_56246891709003
// MI455X (gfx1250) — matrix operand order repaired and checked by reading, not compiled, not run on hardware
//
#include <hip/hip_runtime.h>
#include <hip/hip_bf16.h>
#include <cstddef>

// ---------------------------------------------------------------------------
// CDNA5 WMMA types (wave32): D(16x16,f32) = A(16x32,f16) * B(32x16,f16) + C
// ---------------------------------------------------------------------------
typedef _Float16 half_t;
typedef half_t v16h __attribute__((ext_vector_type(16)));
typedef half_t v8h  __attribute__((ext_vector_type(8)));
typedef float  v8f  __attribute__((ext_vector_type(8)));

static __device__ __forceinline__ v8f wmma_f16(const v16h& a, const v16h& b, const v8f& c) {
    return __builtin_amdgcn_wmma_f32_16x16x32_f16(false, a, false, b, (short)0, c, false, false);
}

// ---------------------------------------------------------------------------
// Implicit-GEMM conv2d on NHWC f16 activations (channels padded to mult. of 32)
// with pre-packed f16 weights [Cpout][KH*KW][Cpin].
// One wave computes 2 output rows x 16 ow x 32 out-channels (4 WMMA tiles).
// A fragment = 2x v8h contiguous loads (b128); B fragment = 1x v16h load.
// OOB handled by pointer clamp + vector zero-select (no branches in K loop).
// ---------------------------------------------------------------------------
template<int KH, int KW>
__global__ __launch_bounds__(32)
void conv2d_nhwc_wmma(const half_t* __restrict__ in, const half_t* __restrict__ wp,
                      const float* __restrict__ bias, const half_t* __restrict__ resid,
                      half_t* __restrict__ out, float* __restrict__ out_nchw,
                      int Cpin, int H, int W, int Cout, int Cpout,
                      int Ho, int Wo, int stride, int pad)
{
    constexpr int KHKW = KH * KW;
    const int lane = threadIdx.x;
    const int row  = lane & 15;
    const bool hi  = lane >= 16;
    const int moff = hi ? 8 : 0;     // M offset for C/D
    const int boff = hi ? 16 : 0;    // K offset for B fragment

    const int wt = Wo >> 4;
    int mt = blockIdx.x;
    const int ow0 = (mt % wt) << 4;  mt /= wt;
    const int mh  = Ho >> 1;
    const int oh0 = (mt % mh) << 1;
    const int b   = mt / mh;
    const int oc0 = blockIdx.y << 5;

    const int ow   = ow0 + row;
    const int iw0  = ow * stride - pad;
    const int ihA  = oh0 * stride - pad;
    const int ihB  = (oh0 + 1) * stride - pad;

    const int ocn0 = oc0 + row;
    const int ocn1 = oc0 + 16 + row;
    const size_t wstride = (size_t)KHKW * Cpin;
    const half_t* __restrict__ wr0 = wp + (size_t)ocn0 * wstride;
    const half_t* __restrict__ wr1 = wp + (size_t)ocn1 * wstride;

    v8f acc00, acc01, acc10, acc11;
    const float bv0 = (ocn0 < Cout) ? bias[ocn0] : 0.0f;
    const float bv1 = (ocn1 < Cout) ? bias[ocn1] : 0.0f;
#pragma unroll
    for (int i = 0; i < 8; ++i) { acc00[i] = bv0; acc01[i] = bv1; acc10[i] = bv0; acc11[i] = bv1; }

    const half_t* __restrict__ inb = in + (size_t)b * H * W * Cpin;
    const v8h zed = {};

#pragma unroll
    for (int cell = 0; cell < KHKW; ++cell) {
        const int kh = cell / KW, kw = cell % KW;
        const int iw  = iw0 + kw;
        const int ih0 = ihA + kh, ih1 = ihB + kh;
        const bool okw = (unsigned)iw  < (unsigned)W;
        const bool ok0 = okw && (unsigned)ih0 < (unsigned)H;
        const bool ok1 = okw && (unsigned)ih1 < (unsigned)H;
        const half_t* p0 = ok0 ? inb + ((size_t)ih0 * W + iw) * Cpin : inb;
        const half_t* p1 = ok1 ? inb + ((size_t)ih1 * W + iw) * Cpin : inb;

        for (int coff = 0; coff < Cpin; coff += 32) {
            v8h r00 = *(const v8h*)(p0 + coff + boff);
            v8h r01 = *(const v8h*)(p0 + coff + 8 + boff);
            v8h r10 = *(const v8h*)(p1 + coff + boff);
            v8h r11 = *(const v8h*)(p1 + coff + 8 + boff);
            r00 = ok0 ? r00 : zed;  r01 = ok0 ? r01 : zed;
            r10 = ok1 ? r10 : zed;  r11 = ok1 ? r11 : zed;

            v16h a0, a1;
#pragma unroll
            for (int e = 0; e < 8; ++e) {
                a0[e] = r00[e]; a0[8 + e] = r01[e];
                a1[e] = r10[e]; a1[8 + e] = r11[e];
            }
            const size_t kb = (size_t)cell * Cpin + coff + boff;
            const v16h b0 = *(const v16h*)(wr0 + kb);
            const v16h b1 = *(const v16h*)(wr1 + kb);

            acc00 = wmma_f16(a0, b0, acc00);
            acc01 = wmma_f16(a0, b1, acc01);
            acc10 = wmma_f16(a1, b0, acc10);
            acc11 = wmma_f16(a1, b1, acc11);
        }
    }

    // C/D layout: N = lane&15 (channel), M = i + (hi?8:0) (pixel along ow)
#pragma unroll
    for (int m = 0; m < 2; ++m) {
        const int oh = oh0 + m;
        const v8f& an0 = m ? acc10 : acc00;
        const v8f& an1 = m ? acc11 : acc01;
#pragma unroll
        for (int i = 0; i < 8; ++i) {
            const int px = ow0 + i + moff;
            const size_t pix = ((size_t)b * Ho + oh) * Wo + px;
            float v0 = an0[i], v1 = an1[i];
            if (resid) {
                v0 += (float)resid[pix * Cpout + ocn0];
                v1 += (float)resid[pix * Cpout + ocn1];
            }
            if (out_nchw) {   // final layer: fp32 NCHW
                if (ocn0 < Cout) out_nchw[((size_t)b * Cout + ocn0) * Ho * Wo + (size_t)oh * Wo + px] = v0;
                if (ocn1 < Cout) out_nchw[((size_t)b * Cout + ocn1) * Ho * Wo + (size_t)oh * Wo + px] = v1;
            } else {
                out[pix * Cpout + ocn0] = (half_t)(ocn0 < Cout ? v0 : 0.0f);
                out[pix * Cpout + ocn1] = (half_t)(ocn1 < Cout ? v1 : 0.0f);
            }
        }
    }
}

// ---------------------------------------------------------------------------
// ConvTranspose2d(k=4,s=2,p=1), Cin==Cout==C (multiple of 32), NHWC f16.
// out[oh,ow,oc] = b + sum_{t=(kh,kw),ic} in[(oh+1-kh)/2,(ow+1-kw)/2,ic]*w[oc][t][ic]
// with invalid-parity taps zero-selected. Weights packed [C][16][C].
// ---------------------------------------------------------------------------
__global__ __launch_bounds__(32)
void convt_nhwc_wmma(const half_t* __restrict__ in, const half_t* __restrict__ wp,
                     const float* __restrict__ bias, half_t* __restrict__ out,
                     int C, int H, int W)
{
    const int Ho = H << 1, Wo = W << 1;
    const int lane = threadIdx.x;
    const int row  = lane & 15;
    const bool hi  = lane >= 16;
    const int moff = hi ? 8 : 0;
    const int boff = hi ? 16 : 0;

    const int wt = Wo >> 4;
    int mt = blockIdx.x;
    const int ow0 = (mt % wt) << 4;  mt /= wt;
    const int mh  = Ho >> 1;
    const int oh0 = (mt % mh) << 1;
    const int b   = mt / mh;
    const int oc0 = blockIdx.y << 5;

    const int ow   = ow0 + row;
    const int ocn0 = oc0 + row;
    const int ocn1 = oc0 + 16 + row;
    const size_t wstride = (size_t)16 * C;
    const half_t* __restrict__ wr0 = wp + (size_t)ocn0 * wstride;
    const half_t* __restrict__ wr1 = wp + (size_t)ocn1 * wstride;

    v8f acc00, acc01, acc10, acc11;
    const float bv0 = bias[ocn0], bv1 = bias[ocn1];
#pragma unroll
    for (int i = 0; i < 8; ++i) { acc00[i] = bv0; acc01[i] = bv1; acc10[i] = bv0; acc11[i] = bv1; }

    const half_t* __restrict__ inb = in + (size_t)b * H * W * C;
    const v8h zed = {};

#pragma unroll
    for (int t = 0; t < 16; ++t) {
        const int kh = t >> 2, kw = t & 3;
        const int tw = ow + 1 - kw;
        const int iw = tw >> 1;
        const bool okw = (tw >= 0) && !(tw & 1) && (iw < W);
        const int th0 = oh0 + 1 - kh,  th1 = oh0 + 2 - kh;
        const int ih0 = th0 >> 1,      ih1 = th1 >> 1;
        const bool ok0 = okw && (th0 >= 0) && !(th0 & 1) && (ih0 < H);
        const bool ok1 = okw && (th1 >= 0) && !(th1 & 1) && (ih1 < H);
        const half_t* p0 = ok0 ? inb + ((size_t)ih0 * W + iw) * C : inb;
        const half_t* p1 = ok1 ? inb + ((size_t)ih1 * W + iw) * C : inb;

        for (int coff = 0; coff < C; coff += 32) {
            v8h r00 = *(const v8h*)(p0 + coff + boff);
            v8h r01 = *(const v8h*)(p0 + coff + 8 + boff);
            v8h r10 = *(const v8h*)(p1 + coff + boff);
            v8h r11 = *(const v8h*)(p1 + coff + 8 + boff);
            r00 = ok0 ? r00 : zed;  r01 = ok0 ? r01 : zed;
            r10 = ok1 ? r10 : zed;  r11 = ok1 ? r11 : zed;

            v16h a0, a1;
#pragma unroll
            for (int e = 0; e < 8; ++e) {
                a0[e] = r00[e]; a0[8 + e] = r01[e];
                a1[e] = r10[e]; a1[8 + e] = r11[e];
            }
            const size_t kb = (size_t)t * C + coff + boff;
            const v16h b0 = *(const v16h*)(wr0 + kb);
            const v16h b1 = *(const v16h*)(wr1 + kb);

            acc00 = wmma_f16(a0, b0, acc00);
            acc01 = wmma_f16(a0, b1, acc01);
            acc10 = wmma_f16(a1, b0, acc10);
            acc11 = wmma_f16(a1, b1, acc11);
        }
    }

#pragma unroll
    for (int m = 0; m < 2; ++m) {
        const int oh = oh0 + m;
        const v8f& an0 = m ? acc10 : acc00;
        const v8f& an1 = m ? acc11 : acc01;
#pragma unroll
        for (int i = 0; i < 8; ++i) {
            const int px = ow0 + i + moff;
            const size_t pix = ((size_t)b * Ho + oh) * Wo + px;
            out[pix * C + ocn0] = (half_t)an0[i];
            out[pix * C + ocn1] = (half_t)an1[i];
        }
    }
}

// ---------------------------------------------------------------------------
// GroupNorm(8, eps=1e-5) + affine + optional SiLU on NHWC f16 (C == Cpad here).
// One 256-thread block per (batch, group); LDS tree reduction, f32 stats.
// ---------------------------------------------------------------------------
__global__ __launch_bounds__(256)
void groupnorm_silu_nhwc(const half_t* __restrict__ in, const float* __restrict__ gamma,
                         const float* __restrict__ beta, half_t* __restrict__ out,
                         int C, int HW, int cgShift, int do_silu)
{
    const int g = blockIdx.x & 7;
    const int b = blockIdx.x >> 3;
    const int Cg = 1 << cgShift;
    const size_t base = (size_t)b * HW * C + (size_t)g * Cg;
    const size_t N = (size_t)HW << cgShift;

    float s = 0.0f, ss = 0.0f;
    for (size_t i = threadIdx.x; i < N; i += 256) {
        const int p = (int)(i >> cgShift);
        const int c = (int)(i & (Cg - 1));
        const float v = (float)in[base + (size_t)p * C + c];
        s += v; ss += v * v;
    }
    __shared__ float r0[256], r1[256];
    r0[threadIdx.x] = s; r1[threadIdx.x] = ss;
    __syncthreads();
    for (int st = 128; st > 0; st >>= 1) {
        if (threadIdx.x < st) {
            r0[threadIdx.x] += r0[threadIdx.x + st];
            r1[threadIdx.x] += r1[threadIdx.x + st];
        }
        __syncthreads();
    }
    const float mean = r0[0] / (float)N;
    const float var  = r1[0] / (float)N - mean * mean;
    const float inv  = rsqrtf(var + 1e-5f);

    for (size_t i = threadIdx.x; i < N; i += 256) {
        const int p = (int)(i >> cgShift);
        const int c = (int)(i & (Cg - 1));
        const int gc = (g << cgShift) + c;
        float y = ((float)in[base + (size_t)p * C + c] - mean) * inv * gamma[gc] + beta[gc];
        if (do_silu) y = y / (1.0f + expf(-y));
        out[base + (size_t)p * C + c] = (half_t)y;
    }
}

// ---------------------------------------------------------------------------
// VQ forward: z_q = codebook[argmin_k ||c_k||^2 - 2 z.c_k]. NHWC f16 latents.
// ---------------------------------------------------------------------------
__global__ __launch_bounds__(256)
void vq_argmin_nhwc(const half_t* __restrict__ ze, const float* __restrict__ cb,
                    half_t* __restrict__ zq, int NP, int K)
{
    const int p = blockIdx.x * 256 + threadIdx.x;
    if (p >= NP) return;
    const half_t* __restrict__ zp = ze + (size_t)p * 64;
    float z[64];
#pragma unroll
    for (int d = 0; d < 64; ++d) z[d] = (float)zp[d];

    int best = 0; float bestd = 3.4e38f;
    for (int k = 0; k < K; ++k) {
        const float* __restrict__ c = cb + (size_t)k * 64;
        float sc = 0.0f;
#pragma unroll
        for (int d = 0; d < 64; ++d) { const float cv = c[d]; sc += cv * (cv - 2.0f * z[d]); }
        if (sc < bestd) { bestd = sc; best = k; }
    }
    const float* __restrict__ c = cb + (size_t)best * 64;
    half_t* __restrict__ o = zq + (size_t)p * 64;
#pragma unroll
    for (int d = 0; d < 64; ++d) o[d] = (half_t)c[d];
}

// ---------------------------------------------------------------------------
// Layout conversion + weight repack kernels (run every call; deterministic).
// ---------------------------------------------------------------------------
__global__ __launch_bounds__(256)
void x_to_nhwc_f16(const float* __restrict__ x, half_t* __restrict__ xh,
                   int C, int Cp, int HW, int total)
{
    const int i = blockIdx.x * 256 + threadIdx.x;
    if (i >= total) return;
    const int c = i % Cp;
    const int rest = i / Cp;
    const int p = rest % HW;
    const int b = rest / HW;
    xh[i] = (half_t)((c < C) ? x[((size_t)b * C + c) * HW + p] : 0.0f);
}

__global__ __launch_bounds__(256)
void repack_conv_w(const float* __restrict__ w, half_t* __restrict__ o,
                   int Cout, int Cin, int KHKW, int Cpin, int total)
{
    const int i = blockIdx.x * 256 + threadIdx.x;
    if (i >= total) return;
    const int ic = i % Cpin;
    const int r  = i / Cpin;
    const int cell = r % KHKW;
    const int oc = r / KHKW;
    float v = 0.0f;
    if (oc < Cout && ic < Cin) v = w[((size_t)oc * Cin + ic) * KHKW + cell];
    o[i] = (half_t)v;
}

__global__ __launch_bounds__(256)
void repack_convt_w(const float* __restrict__ w, half_t* __restrict__ o,
                    int C, int total)   // [Cin][Cout][16] -> [Cout][16][Cin]
{
    const int i = blockIdx.x * 256 + threadIdx.x;
    if (i >= total) return;
    const int ic = i % C;
    const int r  = i / C;
    const int t  = r % 16;
    const int oc = r / 16;
    o[i] = (half_t)w[((size_t)ic * C + oc) * 16 + t];
}

// ---------------------------------------------------------------------------
// Host-side orchestration
// ---------------------------------------------------------------------------
struct CP  { const float* w; const float* b; };
struct GNP { const float* g; const float* b; };
struct PW  { const half_t* w; const float* bias; int Cin, Cpin, Cout, Cpout, khkw; };
struct ResPW { GNP n1; PW c1; GNP n2; PW c2; PW skip; bool has_skip; };

static inline int pad32(int c) { return (c + 31) & ~31; }

static inline void launch_conv(hipStream_t s, const half_t* in, const PW& w,
                               const half_t* resid, half_t* out, float* out_nchw,
                               int B, int H, int W, int k, int stride, int pad)
{
    const int Ho = (H + 2 * pad - k) / stride + 1;
    const int Wo = (W + 2 * pad - k) / stride + 1;
    dim3 grid((Wo / 16) * (Ho / 2) * B, w.Cpout / 32);
    if (k == 3)
        conv2d_nhwc_wmma<3, 3><<<grid, 32, 0, s>>>(in, w.w, w.bias, resid, out, out_nchw,
                                                   w.Cpin, H, W, w.Cout, w.Cpout, Ho, Wo, stride, pad);
    else
        conv2d_nhwc_wmma<1, 1><<<grid, 32, 0, s>>>(in, w.w, w.bias, resid, out, out_nchw,
                                                   w.Cpin, H, W, w.Cout, w.Cpout, Ho, Wo, stride, pad);
}

static inline void launch_convt(hipStream_t s, const half_t* in, const PW& w, half_t* out,
                                int B, int C, int H, int W)
{
    dim3 grid(((W * 2) / 16) * ((H * 2) / 2) * B, C / 32);
    convt_nhwc_wmma<<<grid, 32, 0, s>>>(in, w.w, w.bias, out, C, H, W);
}

static inline void launch_gn(hipStream_t s, const half_t* in, GNP p, half_t* out,
                             int B, int C, int HW, int do_silu)
{
    int cg = C / 8, sh = 0;
    while ((1 << sh) < cg) ++sh;
    groupnorm_silu_nhwc<<<B * 8, 256, 0, s>>>(in, p.g, p.b, out, C, HW, sh, do_silu);
}

static void resblock(hipStream_t s, const ResPW& r, const half_t* I, half_t* O, half_t* T,
                     int B, int cin, int cout, int H, int W)
{
    launch_gn(s, I, r.n1, T, B, cin, H * W, 1);
    launch_conv(s, T, r.c1, nullptr, O, nullptr, B, H, W, 3, 1, 1);
    launch_gn(s, O, r.n2, T, B, cout, H * W, 1);
    const half_t* res;
    if (r.has_skip) {
        launch_conv(s, I, r.skip, nullptr, O, nullptr, B, H, W, 1, 1, 0);
        res = O;   // same-index read-before-write in epilogue: safe
    } else {
        res = I;
    }
    launch_conv(s, T, r.c2, res, O, nullptr, B, H, W, 3, 1, 1);
}

extern "C" void kernel_launch(void* const* d_in, const int* in_sizes, int n_in,
                              void* d_out, int out_size, void* d_ws, size_t ws_size,
                              hipStream_t stream)
{
    (void)in_sizes; (void)n_in; (void)out_size; (void)ws_size;
    const int B = 8;

    // ---- workspace bump allocator ----
    char* wsp = (char*)d_ws;
    auto alloc_h = [&](size_t nelem) -> half_t* {
        half_t* p = (half_t*)wsp;
        wsp += (nelem * sizeof(half_t) + 255) & ~(size_t)255;
        return p;
    };

    // ---- walk inputs in setup_inputs() insertion order ----
    int ix = 0;
    auto F = [&](int i) { return (const float*)d_in[i]; };
    const float* x = F(ix++);
    auto nextC = [&]() { CP p{F(ix), F(ix + 1)}; ix += 2; return p; };
    auto nextG = [&]() { GNP p{F(ix), F(ix + 1)}; ix += 2; return p; };

    auto prep_conv = [&](CP p, int Cout, int Cin, int khkw) {
        PW w; w.Cin = Cin; w.Cpin = pad32(Cin); w.Cout = Cout; w.Cpout = pad32(Cout); w.khkw = khkw;
        const size_t total = (size_t)w.Cpout * khkw * w.Cpin;
        half_t* dst = alloc_h(total);
        repack_conv_w<<<(unsigned)((total + 255) / 256), 256, 0, stream>>>(
            p.w, dst, Cout, Cin, khkw, w.Cpin, (int)total);
        w.w = dst; w.bias = p.b;
        return w;
    };
    auto prep_convt = [&](CP p, int C) {
        PW w; w.Cin = C; w.Cpin = C; w.Cout = C; w.Cpout = C; w.khkw = 16;
        const size_t total = (size_t)C * 16 * C;
        half_t* dst = alloc_h(total);
        repack_convt_w<<<(unsigned)((total + 255) / 256), 256, 0, stream>>>(p.w, dst, C, (int)total);
        w.w = dst; w.bias = p.b;
        return w;
    };
    auto nextR = [&](int cin, int cout) {
        ResPW r;
        r.n1 = nextG(); r.c1 = prep_conv(nextC(), cout, cin, 9);
        r.n2 = nextG(); r.c2 = prep_conv(nextC(), cout, cout, 9);
        r.has_skip = (cin != cout);
        if (r.has_skip) r.skip = prep_conv(nextC(), cout, cin, 1);
        return r;
    };

    // encoder
    PW    e_conv_in = prep_conv(nextC(), 128, 44, 9);
    ResPW e00 = nextR(128, 128), e01 = nextR(128, 128);
    PW    e_down0 = prep_conv(nextC(), 128, 128, 9);
    ResPW e10 = nextR(128, 256), e11 = nextR(256, 256);
    PW    e_down1 = prep_conv(nextC(), 256, 256, 9);
    ResPW e20 = nextR(256, 512), e21 = nextR(512, 512);
    GNP   e_norm = nextG();
    PW    e_cout = prep_conv(nextC(), 64, 512, 9);
    // decoder
    PW    d_conv_in = prep_conv(nextC(), 512, 64, 9);
    ResPW d00 = nextR(512, 512), d01 = nextR(512, 512);
    PW    d_up0 = prep_convt(nextC(), 512);
    ResPW d10 = nextR(512, 256), d11 = nextR(256, 256);
    PW    d_up1 = prep_convt(nextC(), 256);
    ResPW d20 = nextR(256, 128), d21 = nextR(128, 128);
    GNP   d_norm = nextG();
    PW    d_cout = prep_conv(nextC(), 44, 128, 9);
    const float* codebook = F(ix++);

    // ---- activation buffers (NHWC f16, padded channels) ----
    const size_t MAXE = (size_t)8 * 64 * 64 * 256;      // largest activation (elems)
    half_t* A0 = alloc_h(MAXE);
    half_t* A1 = alloc_h(MAXE);
    half_t* A2 = alloc_h(MAXE);
    half_t* XH = alloc_h((size_t)8 * 64 * 64 * 64);     // input, Cpad=64
    half_t* ZE = alloc_h((size_t)8 * 16 * 16 * 64);     // latent z_e

    half_t* cur = A0; half_t* oth = A1; half_t* tmp = A2;
    auto flip = [&]() { half_t* t = cur; cur = oth; oth = t; };

    // ---- input NCHW f32 -> NHWC f16 (channels zero-padded 44 -> 64) ----
    {
        const int total = 8 * 64 * 64 * 64;
        x_to_nhwc_f16<<<(total + 255) / 256, 256, 0, stream>>>(x, XH, 44, 64, 64 * 64, total);
    }

    // =========================== Encoder ===========================
    launch_conv(stream, XH, e_conv_in, nullptr, cur, nullptr, B, 64, 64, 3, 1, 1);

    resblock(stream, e00, cur, oth, tmp, B, 128, 128, 64, 64); flip();
    resblock(stream, e01, cur, oth, tmp, B, 128, 128, 64, 64); flip();
    launch_conv(stream, cur, e_down0, nullptr, oth, nullptr, B, 64, 64, 3, 2, 1); flip();

    resblock(stream, e10, cur, oth, tmp, B, 128, 256, 32, 32); flip();
    resblock(stream, e11, cur, oth, tmp, B, 256, 256, 32, 32); flip();
    launch_conv(stream, cur, e_down1, nullptr, oth, nullptr, B, 32, 32, 3, 2, 1); flip();

    resblock(stream, e20, cur, oth, tmp, B, 256, 512, 16, 16); flip();
    resblock(stream, e21, cur, oth, tmp, B, 512, 512, 16, 16); flip();

    launch_gn(stream, cur, e_norm, tmp, B, 512, 16 * 16, 1);
    launch_conv(stream, tmp, e_cout, nullptr, ZE, nullptr, B, 16, 16, 3, 1, 1);

    // ======================= Vector Quantizer ======================
    vq_argmin_nhwc<<<(8 * 256 + 255) / 256, 256, 0, stream>>>(ZE, codebook, oth, 8 * 256, 512);

    // =========================== Decoder ===========================
    launch_conv(stream, oth, d_conv_in, nullptr, cur, nullptr, B, 16, 16, 3, 1, 1);

    resblock(stream, d00, cur, oth, tmp, B, 512, 512, 16, 16); flip();
    resblock(stream, d01, cur, oth, tmp, B, 512, 512, 16, 16); flip();
    launch_convt(stream, cur, d_up0, oth, B, 512, 16, 16); flip();   // 16 -> 32

    resblock(stream, d10, cur, oth, tmp, B, 512, 256, 32, 32); flip();
    resblock(stream, d11, cur, oth, tmp, B, 256, 256, 32, 32); flip();
    launch_convt(stream, cur, d_up1, oth, B, 256, 32, 32); flip();   // 32 -> 64

    resblock(stream, d20, cur, oth, tmp, B, 256, 128, 64, 64); flip();
    resblock(stream, d21, cur, oth, tmp, B, 128, 128, 64, 64); flip();

    launch_gn(stream, cur, d_norm, tmp, B, 128, 64 * 64, 1);
    launch_conv(stream, tmp, d_cout, nullptr, nullptr, (float*)d_out, B, 64, 64, 3, 1, 1);
}
